// Attention_12240656794051
// MI455X (gfx1250) — matrix operand order repaired and checked by reading, not compiled, not run on hardware
//
#include <hip/hip_runtime.h>
#include <cstdint>

typedef __attribute__((ext_vector_type(16))) __bf16 v16bf;
typedef __attribute__((ext_vector_type(8)))  float  v8f;
typedef __attribute__((ext_vector_type(4)))  unsigned int u32x4;
typedef __attribute__((ext_vector_type(8)))  int i32x8;
typedef __attribute__((ext_vector_type(4)))  int i32x4;

#define HEADS 12
#define C_    768
#define B_    8
#define L_    4096
#define N_    4097
#define M_ALL (B_*N_)   /* 32776 */
#define M_IMG (B_*L_)   /* 32768 */

__device__ __forceinline__ unsigned short f2bfu(float x){
  unsigned u = __float_as_uint(x);
  unsigned r = 0x7FFFu + ((u >> 16) & 1u);
  return (unsigned short)((u + r) >> 16);
}
__device__ __forceinline__ float bf2f(unsigned short h){
  return __uint_as_float(((unsigned)h) << 16);
}
__device__ __forceinline__ v8f wmma_bf16(v16bf a, v16bf b, v8f c){
  return __builtin_amdgcn_wmma_f32_16x16x32_bf16(false, a, false, b, (short)0, c, false, false);
}

// A-operand (16x32 bf16) per ISA 7.12.2: lane<16 -> m=lane, K-pairs {0..7,16..23};
// lane>=16 -> m=lane-16, K-pairs {8..15,24..31}. base points at (m0,k0), ld even.
__device__ __forceinline__ v16bf load_A_frag(const unsigned short* base, int ld, int lane){
  const int m   = lane & 15;
  const int kb2 = (lane & 16) ? 4 : 0;
  const unsigned* row = (const unsigned*)(base + m * ld);
  union { v16bf v; unsigned u[8]; } r;
#pragma unroll
  for (int j = 0; j < 4; ++j) r.u[j] = row[kb2 + j];
#pragma unroll
  for (int j = 4; j < 8; ++j) r.u[j] = row[kb2 + j + 4];
  return r.v;
}
// B-operand (32x16 bf16, K x N row-major): n = lane&15, elements = k in table order
// (lane<16: k 0..7,16..23; lane>=16: k 8..15,24..31).
__device__ __forceinline__ v16bf load_B_frag(const unsigned short* base, int ld, int lane){
  const int n  = lane & 15;
  const int kb = (lane & 16) ? 8 : 0;
  union { v16bf v; unsigned short u[16]; } r;
#pragma unroll
  for (int j = 0; j < 8; ++j){
    r.u[j]     = base[(kb + j) * ld + n];
    r.u[j + 8] = base[(16 + kb + j) * ld + n];
  }
  return r.v;
}

// TDM: load a 64-token x 64-elem bf16 tile (8x8 token window) global->LDS.
// 3D tile: dim0=64 d (contiguous), dim1=8 tokens (stride=row_stride), dim2=8 rows
// (stride=64*row_stride). LDS padding: +1 DWORD every 32 DWORDs -> row stride 66 elems
// (conflict-free WMMA fragment reads).
__device__ __forceinline__ void tdm_load_tile_64tok(
    unsigned lds_off, const unsigned short* gtile, unsigned row_stride)
{
  unsigned long long ga = (unsigned long long)(uintptr_t)gtile;
  unsigned long long s1 = (unsigned long long)row_stride * 64ull;
  u32x4 g0;
  g0[0] = 1u;                                   // count=1, user descriptor
  g0[1] = lds_off;                              // lds_addr
  g0[2] = (unsigned)ga;                         // global_addr[31:0]
  g0[3] = ((unsigned)(ga >> 32) & 0x1FFFFFFu) | (2u << 30);  // addr[56:32], type=2
  i32x8 g1;
  g1[0] = (int)((1u << 16) | (1u << 20) | (4u << 22)); // data_size=2B, pad_en, interval=32dw, amount=1dw
  g1[1] = (int)(64u << 16);                     // tensor_dim0 = 64
  g1[2] = (int)(8u << 16);                      // tensor_dim1 = 8
  g1[3] = (int)(64u << 16);                     // tile_dim0 = 64
  g1[4] = (int)(8u | (8u << 16));               // tile_dim1 = 8, tile_dim2 = 8
  g1[5] = (int)row_stride;                      // tensor_dim0_stride lo32
  g1[6] = (int)((s1 & 0xFFFFull) << 16);        // dim0_stride hi16=0 | dim1_stride lo16
  g1[7] = (int)(s1 >> 16);                      // dim1_stride [47:16]
  i32x4 g2; g2[0] = 8; g2[1] = 0; g2[2] = 0; g2[3] = 0;  // tensor_dim2 = 8
  i32x4 g3 = (i32x4)0;
  i32x8 g4 = (i32x8)0;                          // extra group (clang-23 6-arg form)
  __builtin_amdgcn_tensor_load_to_lds(g0, g1, g2, g3, g4, 0);
}

// ---------------------------------------------------------------------------
// Generic GEMM: C[M,N] = A[M,K]*B[K,N] (+bias). A,B f32 in HBM, converted to
// bf16 in LDS, v_wmma_f32_16x16x32_bf16 compute. 128x128x32 macro tile,
// 8 waves (2x4), double-buffered LDS with register prefetch.
// Requires K%32==0 and N%128==0 (true for all call sites).
// ---------------------------------------------------------------------------
template<bool OUT_BF16, bool HAS_BIAS>
__global__ __launch_bounds__(256) void gemm_bf16_wmma(
    const float* __restrict__ A, const float* __restrict__ Bm,
    const float* __restrict__ bias, void* __restrict__ Cout,
    int M, int N, int K)
{
  __shared__ unsigned short As[2][128][34];   // pad 34: conflict-free A-frag reads
  __shared__ unsigned short Bs[2][32][132];   // pad 132: conflict-free B-frag reads
  const int tid  = threadIdx.x;
  const int lane = tid & 31;
  const int wave = tid >> 5;
  const int wm = wave >> 2;                   // 0..1  (64 rows each)
  const int wn = wave & 3;                    // 0..3  (32 cols each)
  const int m_blk = blockIdx.x * 128;
  const int n_blk = blockIdx.y * 128;

  v8f acc[4][2] = {};
  float4 pA[4], pB[4];
  const int ra = tid >> 3, ca = (tid & 7) * 4;
  const int rb = tid >> 5, cb = lane * 4;

  auto fetch = [&](int k0){
#pragma unroll
    for (int rr = 0; rr < 4; ++rr){
      int gm = m_blk + ra + 32*rr;
      pA[rr] = (gm < M) ? *(const float4*)(A + (size_t)gm * K + k0 + ca)
                        : make_float4(0.f, 0.f, 0.f, 0.f);
    }
#pragma unroll
    for (int rr = 0; rr < 4; ++rr)
      pB[rr] = *(const float4*)(Bm + (size_t)(k0 + rb + 8*rr) * N + n_blk + cb);
  };
  auto stash = [&](int buf){
#pragma unroll
    for (int rr = 0; rr < 4; ++rr){
      int r = ra + 32*rr;
      As[buf][r][ca+0] = f2bfu(pA[rr].x);
      As[buf][r][ca+1] = f2bfu(pA[rr].y);
      As[buf][r][ca+2] = f2bfu(pA[rr].z);
      As[buf][r][ca+3] = f2bfu(pA[rr].w);
    }
#pragma unroll
    for (int rr = 0; rr < 4; ++rr){
      int r = rb + 8*rr;
      Bs[buf][r][cb+0] = f2bfu(pB[rr].x);
      Bs[buf][r][cb+1] = f2bfu(pB[rr].y);
      Bs[buf][r][cb+2] = f2bfu(pB[rr].z);
      Bs[buf][r][cb+3] = f2bfu(pB[rr].w);
    }
  };

  const int nk = K >> 5;
  fetch(0); stash(0);
  __syncthreads();
  for (int kt = 0; kt < nk; ++kt){
    const int buf = kt & 1;
    if (kt + 1 < nk) fetch((kt + 1) << 5);
    v16bf af[4];
#pragma unroll
    for (int i = 0; i < 4; ++i)
      af[i] = load_A_frag(&As[buf][wm*64 + 16*i][0], 34, lane);
    v16bf bfv[2];
#pragma unroll
    for (int j = 0; j < 2; ++j)
      bfv[j] = load_B_frag(&Bs[buf][0][wn*32 + 16*j], 132, lane);
#pragma unroll
    for (int i = 0; i < 4; ++i)
#pragma unroll
      for (int j = 0; j < 2; ++j)
        acc[i][j] = wmma_bf16(af[i], bfv[j], acc[i][j]);
    if (kt + 1 < nk) stash(buf ^ 1);
    __syncthreads();
  }

  const int ncol = lane & 15;
  const int rof  = (lane & 16) ? 8 : 0;
#pragma unroll
  for (int i = 0; i < 4; ++i)
#pragma unroll
    for (int j = 0; j < 2; ++j)
#pragma unroll
      for (int r = 0; r < 8; ++r){
        int gm = m_blk + wm*64 + 16*i + rof + r;
        if (gm < M){
          int gn = n_blk + wn*32 + 16*j + ncol;
          float v = acc[i][j][r];
          if (HAS_BIAS) v += bias[gn];
          if (OUT_BF16) ((unsigned short*)Cout)[(size_t)gm * N + gn] = f2bfu(v);
          else          ((float*)      Cout)[(size_t)gm * N + gn] = v;
        }
      }
}

// ---------------------------------------------------------------------------
// Window attention: one block = one (window, head). 4 waves; wave w owns the
// 16-row q stripe [16w,16w+16). Q/V tiles arrive via TDM (tensor_load_to_lds,
// s_wait_tensorcnt); K is transposed into LDS (Kt[d][key]) so the B-operand of
// S = Q*K^T is a natural row-major K x N tile. Masked softmax uses the
// algebraic identity amask = m_q * m_k (token mask).
// ---------------------------------------------------------------------------
__global__ __launch_bounds__(128) void win_attn(
    const unsigned short* __restrict__ qkv,   // (32776, 2304) bf16: q|k|v per head
    const unsigned short* __restrict__ kvp,   // (32768, 1536) bf16: k_pre|v_pre
    const float* __restrict__ mask,           // (B, L, C) token mask (read col 0)
    float* __restrict__ attnbuf)              // (32776, 768) f32
{
  __shared__ unsigned short Qs[64*66];        // TDM-padded stride 66
  __shared__ unsigned short Vs[128*66];       // keys 0..63 = prefix, 64..127 = img
  __shared__ unsigned short Kt[64*132];       // K^T: [d][key], pad 132
  __shared__ unsigned short Ps[64*132];       // softmaxed probs (bf16)
  __shared__ float msk[64];

  const int tid = threadIdx.x, lane = tid & 31, wave = tid >> 5;
  const int head = blockIdx.x % HEADS;
  const int w    = blockIdx.x / HEADS;
  const int b    = w >> 6;
  const int hgi  = (w >> 3) & 7;
  const int wgi  = w & 7;
  const int y0 = hgi * 8, x0 = wgi * 8;
  const int l0 = y0 * 64 + x0;

  if (wave == 0){
    const size_t qrow = (size_t)(b * N_ + 1 + l0);
    tdm_load_tile_64tok((unsigned)(size_t)(void*)Qs,
                        qkv + qrow * 2304 + head * 64, 2304);
    tdm_load_tile_64tok((unsigned)(size_t)(void*)Vs,
                        kvp + (size_t)(b * L_ + l0) * 1536 + 768 + head * 64, 1536);
    tdm_load_tile_64tok((unsigned)(size_t)(void*)(Vs + 64*66),
                        qkv + qrow * 2304 + 1536 + head * 64, 2304);
  }
  { // manual transposed K fill: thread = key row (0..63 prefix, 64..127 img)
    const int t = tid & 63;
    const int l = (y0 + (t >> 3)) * 64 + (x0 + (t & 7));
    const unsigned short* krow = (tid < 64)
        ? (kvp + (size_t)(b * L_ + l) * 1536 + head * 64)
        : (qkv + (size_t)(b * N_ + 1 + l) * 2304 + 768 + head * 64);
#pragma unroll
    for (int d = 0; d < 64; d += 8){
      uint4 pk = *(const uint4*)(krow + d);
      const unsigned short* pp = (const unsigned short*)&pk;
#pragma unroll
      for (int e = 0; e < 8; ++e) Kt[(d + e) * 132 + tid] = pp[e];
    }
    if (tid < 64) msk[tid] = mask[(size_t)(b * L_ + l) * C_];
  }
  if (wave == 0) __builtin_amdgcn_s_wait_tensorcnt(0);
  __syncthreads();

  // S stripe: 16 q rows x 128 keys = 8 accumulators, K-dim 64 = 2 wmma steps
  v8f accS[8] = {};
#pragma unroll
  for (int kk = 0; kk < 64; kk += 32){
    v16bf a = load_A_frag(Qs + (wave*16)*66 + kk, 66, lane);
#pragma unroll
    for (int nt = 0; nt < 8; ++nt){
      v16bf bb = load_B_frag(Kt + kk*132 + nt*16, 132, lane);
      accS[nt] = wmma_bf16(a, bb, accS[nt]);
    }
  }
  // masked softmax; C-layout: lane<16 rows r, lane>=16 rows 8+r; n = lane&15
  const int ncol = lane & 15;
  const int rof  = (lane & 16) ? 8 : 0;
  float mqv[8], rmax[8];
#pragma unroll
  for (int r = 0; r < 8; ++r){ mqv[r] = msk[wave*16 + rof + r]; rmax[r] = -3.0e38f; }
#pragma unroll
  for (int nt = 0; nt < 8; ++nt){
    const float mk = msk[(nt*16 + ncol) & 63];   // prefix & img keys share token mask
#pragma unroll
    for (int r = 0; r < 8; ++r){
      float v = accS[nt][r] * 0.125f;            // hd^-0.5
      v = (mqv[r] * mk != 0.f) ? v : -10000.f;
      accS[nt][r] = v;
      rmax[r] = fmaxf(rmax[r], v);
    }
  }
#pragma unroll
  for (int off = 1; off < 16; off <<= 1)
#pragma unroll
    for (int r = 0; r < 8; ++r)
      rmax[r] = fmaxf(rmax[r], __shfl_xor(rmax[r], off, 32));
  float rsum[8] = {};
#pragma unroll
  for (int nt = 0; nt < 8; ++nt)
#pragma unroll
    for (int r = 0; r < 8; ++r){
      float p = __expf(accS[nt][r] - rmax[r]);
      accS[nt][r] = p;
      rsum[r] += p;
    }
#pragma unroll
  for (int off = 1; off < 16; off <<= 1)
#pragma unroll
    for (int r = 0; r < 8; ++r)
      rsum[r] += __shfl_xor(rsum[r], off, 32);
  // spill P (unnormalized) to wave-private LDS stripe, reload as A-operand
#pragma unroll
  for (int nt = 0; nt < 8; ++nt)
#pragma unroll
    for (int r = 0; r < 8; ++r)
      Ps[(wave*16 + rof + r)*132 + nt*16 + ncol] = f2bfu(accS[nt][r]);
  asm volatile("s_wait_dscnt 0x0" ::: "memory");   // wave-local LDS RAW

  // O = P * V : 16 x 64, K-dim 128 = 4 wmma steps
  v8f accO[4] = {};
#pragma unroll
  for (int kk = 0; kk < 128; kk += 32){
    v16bf a = load_A_frag(Ps + (wave*16)*132 + kk, 132, lane);
#pragma unroll
    for (int dt = 0; dt < 4; ++dt){
      v16bf bb = load_B_frag(Vs + kk*66 + dt*16, 66, lane);
      accO[dt] = wmma_bf16(a, bb, accO[dt]);
    }
  }
#pragma unroll
  for (int r = 0; r < 8; ++r){
    const int t = wave*16 + rof + r;
    const int l = (y0 + (t >> 3)) * 64 + (x0 + (t & 7));
    const size_t o = (size_t)(b * N_ + 1 + l) * C_ + head * 64;
    const float inv = 1.0f / rsum[r];
#pragma unroll
    for (int dt = 0; dt < 4; ++dt)
      attnbuf[o + dt*16 + ncol] = accO[dt][r] * inv;
  }
}

// ---------------------------------------------------------------------------
// Global (cls) attention: one block per (b, head). 1 query x 4096 keys.
// Tiny (0.05 GFLOP) -> plain VALU with LDS logits.
// ---------------------------------------------------------------------------
__global__ __launch_bounds__(256) void global_attn(
    const unsigned short* __restrict__ qkv,
    const float* __restrict__ mask,
    const float* __restrict__ gmask,          // (B, 12, 1, L)
    float* __restrict__ xcls,                 // (8, 768) f32
    float* __restrict__ attnbuf)
{
  __shared__ float qs[64];
  __shared__ float lg[4096];
  __shared__ float red[256];
  const int tid = threadIdx.x;
  const int b = blockIdx.x / HEADS, h = blockIdx.x % HEADS;

  if (tid < 64)
    qs[tid] = bf2f(qkv[(size_t)(b * N_) * 2304 + h * 64 + tid]);
  __syncthreads();

  float lmax = -3.0e38f;
  for (int j = tid; j < L_; j += 256){
    const unsigned short* kr = qkv + (size_t)(b * N_ + 1 + j) * 2304 + 768 + h * 64;
    float dot = 0.f;
#pragma unroll
    for (int d = 0; d < 64; ++d) dot += qs[d] * bf2f(kr[d]);
    // amask_g = gmask * key_strength, key_strength == token mask
    float am = gmask[(size_t)(b * HEADS + h) * L_ + j] * mask[(size_t)(b * L_ + j) * C_];
    float v = (am != 0.f) ? dot * 0.125f : -10000.f;
    lg[j] = v;
    lmax = fmaxf(lmax, v);
  }
  red[tid] = lmax; __syncthreads();
  for (int s = 128; s > 0; s >>= 1){ if (tid < s) red[tid] = fmaxf(red[tid], red[tid+s]); __syncthreads(); }
  const float gmax = red[0];
  __syncthreads();
  float lsum = 0.f;
  for (int j = tid; j < L_; j += 256){ float p = __expf(lg[j] - gmax); lg[j] = p; lsum += p; }
  red[tid] = lsum; __syncthreads();
  for (int s = 128; s > 0; s >>= 1){ if (tid < s) red[tid] += red[tid+s]; __syncthreads(); }
  const float inv = 1.0f / red[0];
  __syncthreads();
  if (tid < 64){
    const unsigned short* vb = qkv + (size_t)(b * N_ + 1) * 2304 + 1536 + h * 64 + tid;
    float acc = 0.f;
    for (int j = 0; j < L_; ++j) acc += lg[j] * bf2f(vb[(size_t)j * 2304]);
    const float o = acc * inv;
    xcls[b * C_ + h * 64 + tid] = o;
    attnbuf[(size_t)(b * N_) * C_ + h * 64 + tid] = o;
  }
}

// attn_b = softmax over a single key == 1.0 identically -> x_img += v_cls broadcast
__global__ __launch_bounds__(256) void add_vcls(
    float* __restrict__ attnbuf, const unsigned short* __restrict__ kvg)
{
  const size_t idx = (size_t)blockIdx.x * 256 + threadIdx.x;
  if (idx >= (size_t)B_ * L_ * C_) return;
  const int c = (int)(idx % C_);
  const size_t bl = idx / C_;
  const int l = (int)(bl % L_);
  const int b = (int)(bl / L_);
  attnbuf[(size_t)(b * N_ + 1 + l) * C_ + c] += bf2f(kvg[b * 1536 + 768 + c]);
}

// ---------------------------------------------------------------------------
extern "C" void kernel_launch(void* const* d_in, const int* in_sizes, int n_in,
                              void* d_out, int out_size, void* d_ws, size_t ws_size,
                              hipStream_t stream)
{
  const float* x      = (const float*)d_in[0];
  const float* mask   = (const float*)d_in[1];
  const float* gmask  = (const float*)d_in[2];
  const float* lp     = (const float*)d_in[3];
  const float* w_qkv  = (const float*)d_in[4];
  const float* w_kvp  = (const float*)d_in[5];
  const float* w_kvg  = (const float*)d_in[6];
  const float* w_proj = (const float*)d_in[7];
  const float* b_proj = (const float*)d_in[8];
  (void)in_sizes; (void)n_in; (void)out_size; (void)ws_size;

  char* ws = (char*)d_ws;
  unsigned short* qkv  = (unsigned short*)ws;               size_t o = (size_t)M_ALL * 2304 * 2;
  unsigned short* kvp  = (unsigned short*)(ws + o);         o += (size_t)M_IMG * 1536 * 2;
  float*          attn = (float*)(ws + o);                  o += (size_t)M_ALL * C_ * 4;
  float*          xcls = (float*)(ws + o);                  o += (size_t)B_ * C_ * 4;
  unsigned short* kvg  = (unsigned short*)(ws + o);

  // 1) qkv = x @ w_qkv  (bf16 out)
  gemm_bf16_wmma<true, false><<<dim3(257, 18), 256, 0, stream>>>(
      x, w_qkv, nullptr, qkv, M_ALL, 2304, C_);
  // 2) kv_prefix = layout_prefix @ w_kv_prefix  (bf16 out)
  gemm_bf16_wmma<true, false><<<dim3(256, 12), 256, 0, stream>>>(
      lp, w_kvp, nullptr, kvp, M_IMG, 1536, C_);
  // 3) window attention -> attn rows (img)
  win_attn<<<512 * HEADS, 128, 0, stream>>>(qkv, kvp, mask, attn);
  // 4) cls attention -> attn row 0 per batch + xcls
  global_attn<<<B_ * HEADS, 256, 0, stream>>>(qkv, mask, gmask, xcls, attn);
  // 5) kv_global = x_cls @ w_kv_global  (bf16 out; only v half consumed)
  gemm_bf16_wmma<true, false><<<dim3(1, 12), 256, 0, stream>>>(
      xcls, w_kvg, nullptr, kvg, B_, 1536, C_);
  // 6) x_img += v_cls (attn_b == 1 identically)
  add_vcls<<<(B_ * L_ * C_ + 255) / 256, 256, 0, stream>>>(attn, kvg);
  // 7) out = attn @ w_proj + b_proj  (f32 out)
  gemm_bf16_wmma<false, true><<<dim3(257, 6), 256, 0, stream>>>(
      attn, w_proj, b_proj, d_out, M_ALL, C_, C_);
}
